// RNN_Net_57002805953205
// MI455X (gfx1250) — compile-verified, matrix operand order checked by reading, not run on hardware
//
#include <hip/hip_runtime.h>
#include <hip/hip_bf16.h>

#define HIDDEN 128
#define SEQ 41
#define BATCH 8192

#define WAVES 4
#define ROWS_PER_BLOCK 32                        // block owns 32 batch rows; waves split N-tiles
#define HPAD 136                                 // h row stride in halves (128 + 8 pad -> 272B, 16B aligned)
#define NT 24                                    // 384 / 16 N-tiles (r,z,n stacked)
#define KTS 4                                    // 128 / 32 K-steps

typedef __attribute__((ext_vector_type(16))) _Float16 v16h;
typedef __attribute__((ext_vector_type(8)))  float    v8f;

union FragU {
    uint4 u[2];   // 32 bytes
    v16h  h;
};

__device__ __forceinline__ float fast_sigmoid(float v) {
    // 1/(1+exp(-v)) using v_exp_f32 (base-2) + v_rcp_f32
    return __builtin_amdgcn_rcpf(1.0f + __builtin_amdgcn_exp2f(v * -1.4426950408889634f));
}
__device__ __forceinline__ float fast_tanh(float v) {
    // tanh(v) = 2*sigmoid(2v) - 1
    return 2.0f * __builtin_amdgcn_rcpf(1.0f + __builtin_amdgcn_exp2f(v * -2.8853900817779268f)) - 1.0f;
}

__global__ __launch_bounds__(WAVES * 32) void gru_wmma_kernel(
    const float* __restrict__ x,        // [B][SEQ][1]
    const float* __restrict__ hidden0,  // [1][B][H]
    const float* __restrict__ w_ih,     // [3H][1]
    const float* __restrict__ w_hh,     // [3H][H]
    const float* __restrict__ b_ih,     // [3H]
    const float* __restrict__ b_hh,     // [3H]
    const float* __restrict__ fc_w,     // [1][H]
    const float* __restrict__ fc_b,     // [1]
    const float* __restrict__ prelu_w,  // [1]
    float* __restrict__ out)            // [B][1]
{
    // w_hh^T as f16 WMMA B-fragments: 24 N-tiles x 4 K-steps, 1KB each = 96KB.
    __shared__ __align__(16) uint4 Bfrag[NT * KTS * 64];
    // Double-buffered GRU hidden state (f16): read buf p, write buf 1-p -> one barrier/step.
    __shared__ __align__(16) _Float16 Hst[2][ROWS_PER_BLOCK][HPAD];   // ~17 KB
    __shared__ float Xst[2][ROWS_PER_BLOCK];
    __shared__ float WI[3 * HIDDEN], BI[3 * HIDDEN], BH[3 * HIDDEN], FCW[HIDDEN];

    const int tid  = threadIdx.x;
    const int lane = tid & 31;
    const int wave = tid >> 5;
    const int lmod = lane & 15;      // N (or M-row) within tile
    const int lhi  = lane >> 4;      // K-octet / M-half selector
    const int rowBase = blockIdx.x * ROWS_PER_BLOCK;

    // ---------------- one-time staging ----------------
    // Fragment layout per 16x16x32 f16 operand striping:
    //   lane L holds 16 halves; half hh -> k_local = hh + 8*((hh>>3) + (L>>4)), n = L%16.
    //   Stored so each lane reads 2 contiguous uint4 (conflict-free 512B runs).
    {
        _Float16* Bh = (_Float16*)Bfrag;
        for (int e = tid; e < NT * KTS * 32 * 16; e += blockDim.x) {
            int jn = e >> 11;            // / 2048
            int r  = e & 2047;
            int kt = r >> 9;             // / 512
            int r2 = r & 511;
            int L  = r2 >> 4;
            int hh = r2 & 15;
            int n  = jn * 16 + (L & 15);
            int k  = kt * 32 + hh + 8 * ((hh >> 3) + (L >> 4));
            int off = (jn * KTS + kt) * 512 +
                      ((hh < 8) ? (L * 8 + hh) : (256 + L * 8 + (hh - 8)));
            Bh[off] = (_Float16)w_hh[n * HIDDEN + k];
        }
    }
    for (int i = tid; i < 3 * HIDDEN; i += blockDim.x) {
        WI[i] = w_ih[i]; BI[i] = b_ih[i]; BH[i] = b_hh[i];
    }
    for (int i = tid; i < HIDDEN; i += blockDim.x) FCW[i] = fc_w[i];
    // hidden state init into buffer 0
    for (int i = tid; i < ROWS_PER_BLOCK * HIDDEN; i += blockDim.x) {
        int r = i >> 7, c = i & 127;
        Hst[0][r][c] = (_Float16)hidden0[(rowBase + r) * HIDDEN + c];
    }
    if (tid < ROWS_PER_BLOCK) Xst[0][tid] = x[(rowBase + tid) * SEQ + 0];
    __syncthreads();

    const int jt0 = 2 * wave;            // this wave's two hidden-column tiles

    // ---------------- sequential time loop: one barrier per step ----------------
    for (int t = 0; t < SEQ; ++t) {
        const int p = t & 1;

        // prefetch-stage x for t+1 into the other buffer (readers sync at next barrier)
        if (wave == 0 && t + 1 < SEQ)
            Xst[1 - p][lane] = x[(rowBase + lane) * SEQ + (t + 1)];

        // x values in C/D layout: row m = mt*16 + v + 8*lhi
        float xv[2][8];
        #pragma unroll
        for (int mt = 0; mt < 2; ++mt)
            #pragma unroll
            for (int v = 0; v < 8; ++v)
                xv[mt][v] = Xst[p][mt * 16 + v + 8 * lhi];

        // A fragments (h(t) as f16) : 2 M-tiles x 4 K-steps, register-resident
        FragU A[2][KTS];
        #pragma unroll
        for (int mt = 0; mt < 2; ++mt) {
            const _Float16* rowp = &Hst[p][mt * 16 + lmod][0];
            #pragma unroll
            for (int kt = 0; kt < KTS; ++kt) {
                A[mt][kt].u[0] = *(const uint4*)(rowp + kt * 32 + 8 * lhi);
                A[mt][kt].u[1] = *(const uint4*)(rowp + kt * 32 + 16 + 8 * lhi);
            }
        }

        // this wave's 2 hidden-column tiles (disjoint h columns across waves)
        #pragma unroll
        for (int jj = 0; jj < 2; ++jj) {
            const int jt = jt0 + jj;
            const int j = jt * 16 + lmod;
            const float wi_r = WI[j], wi_z = WI[HIDDEN + j], wi_n = WI[2 * HIDDEN + j];
            const float bi_r = BI[j], bi_z = BI[HIDDEN + j], bi_n = BI[2 * HIDDEN + j];
            const float bh_r = BH[j], bh_z = BH[HIDDEN + j], bh_n = BH[2 * HIDDEN + j];

            v8f accr[2] = {}, accz[2] = {}, accn[2] = {};

            #pragma unroll
            for (int kt = 0; kt < KTS; ++kt) {
                FragU Br, Bz, Bn;
                const uint4* fr = &Bfrag[((jt)      * KTS + kt) * 64];
                const uint4* fz = &Bfrag[((jt + 8)  * KTS + kt) * 64];
                const uint4* fn = &Bfrag[((jt + 16) * KTS + kt) * 64];
                Br.u[0] = fr[lane]; Br.u[1] = fr[32 + lane];
                Bz.u[0] = fz[lane]; Bz.u[1] = fz[32 + lane];
                Bn.u[0] = fn[lane]; Bn.u[1] = fn[32 + lane];
                #pragma unroll
                for (int mt = 0; mt < 2; ++mt) {
                    accr[mt] = __builtin_amdgcn_wmma_f32_16x16x32_f16(
                        false, A[mt][kt].h, false, Br.h, (short)0, accr[mt], false, false);
                    accz[mt] = __builtin_amdgcn_wmma_f32_16x16x32_f16(
                        false, A[mt][kt].h, false, Bz.h, (short)0, accz[mt], false, false);
                    accn[mt] = __builtin_amdgcn_wmma_f32_16x16x32_f16(
                        false, A[mt][kt].h, false, Bn.h, (short)0, accn[mt], false, false);
                }
            }

            // elementwise GRU update in C/D register layout; h(t+1) -> buffer 1-p
            #pragma unroll
            for (int mt = 0; mt < 2; ++mt) {
                #pragma unroll
                for (int v = 0; v < 8; ++v) {
                    const float ghr = accr[mt][v] + bh_r;
                    const float ghz = accz[mt][v] + bh_z;
                    const float ghn = accn[mt][v] + bh_n;
                    const float gxr = xv[mt][v] * wi_r + bi_r;
                    const float gxz = xv[mt][v] * wi_z + bi_z;
                    const float gxn = xv[mt][v] * wi_n + bi_n;
                    const float rg = fast_sigmoid(gxr + ghr);
                    const float zg = fast_sigmoid(gxz + ghz);
                    const float ng = fast_tanh(gxn + rg * ghn);
                    const int row = mt * 16 + v + 8 * lhi;
                    const float hold = (float)Hst[p][row][j];
                    const float hnew = (1.0f - zg) * ng + zg * hold;
                    Hst[1 - p][row][j] = (_Float16)hnew;
                }
            }
        }
        __syncthreads();   // h(t+1), x(t+1) complete & visible before next step
    }

    // ---------------- FC(128->1) + PReLU; final h is in buffer SEQ&1 ----------------
    if (tid < ROWS_PER_BLOCK) {
        float s = fc_b[0];
        #pragma unroll 8
        for (int c = 0; c < HIDDEN; ++c)
            s += (float)Hst[SEQ & 1][tid][c] * FCW[c];
        const float pw = prelu_w[0];
        out[rowBase + tid] = (s >= 0.0f) ? s : pw * s;
    }
}

extern "C" void kernel_launch(void* const* d_in, const int* in_sizes, int n_in,
                              void* d_out, int out_size, void* d_ws, size_t ws_size,
                              hipStream_t stream) {
    (void)in_sizes; (void)n_in; (void)d_ws; (void)ws_size; (void)out_size;
    const float* x       = (const float*)d_in[0];
    const float* hidden0 = (const float*)d_in[1];
    const float* w_ih    = (const float*)d_in[2];
    const float* w_hh    = (const float*)d_in[3];
    const float* b_ih    = (const float*)d_in[4];
    const float* b_hh    = (const float*)d_in[5];
    const float* fc_w    = (const float*)d_in[6];
    const float* fc_b    = (const float*)d_in[7];
    const float* prelu_w = (const float*)d_in[8];
    float* out = (float*)d_out;

    dim3 grid(BATCH / ROWS_PER_BLOCK);   // 256 blocks
    dim3 block(WAVES * 32);              // 128 threads = 4 waves
    gru_wmma_kernel<<<grid, block, 0, stream>>>(
        x, hidden0, w_ih, w_hh, b_ih, b_hh, fc_w, fc_b, prelu_w, out);
}
